// Net_53283364274294
// MI455X (gfx1250) — compile-verified, matrix operand order checked by reading, not run on hardware
//
#include <hip/hip_runtime.h>
#include <hip/hip_bf16.h>

// ---------------- problem constants ----------------
#define NN      100000      // nodes
#define NE      400000      // edges
#define NG      4000        // graphs
#define DD      70          // hidden dim
#define K12     840         // 12*D
#define NKC     27          // k-chunks of 32 (840 padded to 864)
#define NNT     5           // 5 column tiles of 16 cover 70 (padded to 80)
#define NLAYER  4
#define AVG_LOG 1.2465355243460257f
#define INV_BN  0.9999950000374997f   // 1/sqrt(1+1e-5)

// wpack strides (ushort elements)
#define WKC_STRIDE   512                // 32 lanes * 16 elems per k-chunk
#define WNT_STRIDE   (NKC * WKC_STRIDE) // per column-tile
#define WL_STRIDE    (NNT * WNT_STRIDE) // per layer

typedef __attribute__((ext_vector_type(16))) __bf16 v16bf;
typedef __attribute__((ext_vector_type(8)))  float  v8f;

__constant__ int c_atom_off[9] = {0, 119, 124, 136, 148, 158, 164, 170, 172};

static __device__ __forceinline__ unsigned short f2bf(float f) {
    unsigned int u = __float_as_uint(f);
    unsigned int r = (u + 0x7fffu + ((u >> 16) & 1u)) >> 16;   // RNE
    return (unsigned short)r;
}

// ---------------- generic zero ----------------
__global__ void k_zero_f(float* __restrict__ p, int n) {
    for (int i = blockIdx.x * blockDim.x + threadIdx.x; i < n; i += gridDim.x * blockDim.x)
        p[i] = 0.0f;
}

// ---------------- per-layer reduction-buffer init ----------------
__global__ void k_init_layer(float* __restrict__ s, float* __restrict__ s2,
                             int* __restrict__ mn, int* __restrict__ mx, int n) {
    for (int i = blockIdx.x * blockDim.x + threadIdx.x; i < n; i += gridDim.x * blockDim.x) {
        s[i]  = 0.0f;
        s2[i] = 0.0f;
        mn[i] = 0x7f800000;   // +inf bits; messages are >= 0 so int order == float order
        mx[i] = 0;            // 0.0f bits
    }
}

// ---------------- in-degree count ----------------
__global__ void k_count(const int* __restrict__ ei, float* __restrict__ cnt) {
    for (int e = blockIdx.x * blockDim.x + threadIdx.x; e < NE; e += gridDim.x * blockDim.x)
        atomicAdd(&cnt[ei[NE + e]], 1.0f);
}

__global__ void k_amp(const float* __restrict__ cnt, float* __restrict__ amp) {
    for (int n = blockIdx.x * blockDim.x + threadIdx.x; n < NN; n += gridDim.x * blockDim.x) {
        float cs = fmaxf(cnt[n], 1.0f);
        amp[n] = __logf(cs + 1.0f) * (1.0f / AVG_LOG);
    }
}

// ---------------- pre-pack post_w into WMMA B fragments (bf16) ----------------
// wpack[l*WL_STRIDE + nt*WNT_STRIDE + kc*WKC_STRIDE + lane*16 + e]
// lane: n = nt*16 + (lane&15), half = lane>>4 ; elem e: K = kc*32 + 16*(e>>3) + 8*half + (e&7)
// (mirrors the 16-bit A-matrix K striping from CDNA5 ISA §7.12.2)
__global__ void k_wpack(const float* __restrict__ post_w, unsigned short* __restrict__ wpack) {
    const int total = NLAYER * WL_STRIDE;
    for (int id = blockIdx.x * blockDim.x + threadIdx.x; id < total; id += gridDim.x * blockDim.x) {
        int e    = id & 15;
        int lane = (id >> 4) & 31;
        int kc   = (id >> 9) % NKC;
        int nt   = (id / WNT_STRIDE) % NNT;
        int l    = id / WL_STRIDE;
        int n    = nt * 16 + (lane & 15);
        int half = lane >> 4;
        int K    = kc * 32 + 16 * (e >> 3) + 8 * half + (e & 7);
        float v  = (K < K12 && n < DD) ? post_w[(l * K12 + K) * DD + n] : 0.0f;
        wpack[id] = f2bf(v);
    }
}

// ---------------- AtomEncoder: h = sum of 9 embedding rows ----------------
__global__ void k_atom(const int* __restrict__ x, const float* __restrict__ emb,
                       float* __restrict__ h) {
    for (int id = blockIdx.x * blockDim.x + threadIdx.x; id < NN * DD;
         id += gridDim.x * blockDim.x) {
        int n = id / DD, d = id - n * DD;
        float acc = 0.0f;
#pragma unroll
        for (int f = 0; f < 9; ++f)
            acc += emb[(x[n * 9 + f] + c_atom_off[f]) * DD + d];
        h[id] = acc;
    }
}

// ---------------- edge messages + segment reductions (atomics) ----------------
__global__ void k_edge(const int* __restrict__ ei, const float* __restrict__ ea,
                       const float* __restrict__ h, float* __restrict__ s,
                       float* __restrict__ s2, int* __restrict__ mn, int* __restrict__ mx) {
    for (int id = blockIdx.x * blockDim.x + threadIdx.x; id < NE * DD;
         id += gridDim.x * blockDim.x) {
        int e = id / DD, d = id - e * DD;
        int src = ei[e], dst = ei[NE + e];
        float m = h[src * DD + d] + h[dst * DD + d] + ea[id];
        m = fmaxf(m, 0.0f);
        int a = dst * DD + d;
        atomicAdd(&s[a], m);
        atomicAdd(&s2[a], m * m);
        int bits = __float_as_int(m);       // m >= 0 -> int order == float order
        atomicMin(&mn[a], bits);
        atomicMax(&mx[a], bits);
    }
}

// ---------------- node kernel: agg build + bf16 WMMA GEMM + BN/ReLU/residual ----
// One wave (32 threads) per block; wave owns a 32-node tile (two 16-row WMMA tiles
// sharing the same B fragments -> 10 back-to-back WMMAs per k-step, one wait).
__global__ void __launch_bounds__(32)
k_node(const float* __restrict__ s, const float* __restrict__ s2,
       const int* __restrict__ mnb, const int* __restrict__ mxb,
       const float* __restrict__ cnt, const float* __restrict__ amp,
       const unsigned short* __restrict__ wpack,
       const float* __restrict__ post_b, const float* __restrict__ gamma,
       const float* __restrict__ beta, float* __restrict__ h, int layer) {
    __shared__ unsigned short A[32][880];   // 56,320 B; 880-stride staggers banks

    const int lane  = threadIdx.x & 31;
    const int node0 = blockIdx.x * 32;

    // ---- phase 1: build 840-wide agg rows (bf16) in LDS for 32 nodes ----
    for (int idx = lane; idx < 32 * DD; idx += 32) {
        int mrow = idx / DD, d = idx - mrow * DD;
        int node = node0 + mrow;
        float meanv = 0.0f, mnv = 0.0f, mxv = 0.0f, stdv = 0.0f, av = 1.0f;
        if (node < NN) {
            float c  = cnt[node];
            float cs = fmaxf(c, 1.0f);
            int a = node * DD + d;
            float sv = s[a], s2v = s2[a];
            meanv = sv / cs;
            float var = fmaxf(s2v / cs - meanv * meanv, 0.0f);
            stdv = sqrtf(var + 1e-5f);
            if (c > 0.0f) { mnv = __int_as_float(mnb[a]); mxv = __int_as_float(mxb[a]); }
            av = amp[node];
        }
        float ia = 1.0f / av;
        A[mrow][d]           = f2bf(meanv);
        A[mrow][DD + d]      = f2bf(mnv);
        A[mrow][2 * DD + d]  = f2bf(mxv);
        A[mrow][3 * DD + d]  = f2bf(stdv);
        A[mrow][4 * DD + d]  = f2bf(meanv * av);
        A[mrow][5 * DD + d]  = f2bf(mnv * av);
        A[mrow][6 * DD + d]  = f2bf(mxv * av);
        A[mrow][7 * DD + d]  = f2bf(stdv * av);
        A[mrow][8 * DD + d]  = f2bf(meanv * ia);
        A[mrow][9 * DD + d]  = f2bf(mnv * ia);
        A[mrow][10 * DD + d] = f2bf(mxv * ia);
        A[mrow][11 * DD + d] = f2bf(stdv * ia);
    }
    // zero K-pad 840..879 (so pad never injects NaN into WMMA)
    for (int idx = lane; idx < 32 * 40; idx += 32) {
        int mrow = idx / 40;
        A[mrow][K12 + (idx - mrow * 40)] = 0;
    }
    __syncthreads();   // single-wave WG: DS in-order anyway; compiler orders waits

    // ---- phase 2: 27 k-steps x (2 row tiles x 5 col tiles) bf16 WMMA ----
    const int half = lane >> 4;
    const int mln  = lane & 15;
    const unsigned short* wbase = wpack + (size_t)layer * WL_STRIDE + lane * 16;

    v8f acc[2][NNT] = {};
    for (int kc = 0; kc < NKC; ++kc) {
        // prefetch next k-chunk's B block (global_prefetch_b8)
        if (kc + 1 < NKC)
            __builtin_prefetch(wbase + (kc + 1) * WKC_STRIDE, 0, 0);

        // A fragments: two 16-row tiles from LDS, two contiguous 16B runs each
        v16bf a0, a1;
        {
            const unsigned short* rp0 = &A[mln][kc * 32 + 8 * half];
            const unsigned short* rp1 = &A[16 + mln][kc * 32 + 8 * half];
#pragma unroll
            for (int e = 0; e < 8; ++e) {
                a0[e]     = __builtin_bit_cast(__bf16, rp0[e]);
                a0[8 + e] = __builtin_bit_cast(__bf16, rp0[16 + e]);
                a1[e]     = __builtin_bit_cast(__bf16, rp1[e]);
                a1[8 + e] = __builtin_bit_cast(__bf16, rp1[16 + e]);
            }
        }
        // B fragments: all 5 column tiles into distinct registers (one wait, then
        // 10 back-to-back WMMAs). Constant offsets -> immediate ioffset loads.
        v16bf b[NNT];
        const unsigned short* wk = wbase + kc * WKC_STRIDE;
#pragma unroll
        for (int nt = 0; nt < NNT; ++nt)
#pragma unroll
            for (int e = 0; e < 16; ++e)
                b[nt][e] = __builtin_bit_cast(__bf16, wk[nt * WNT_STRIDE + e]);

#pragma unroll
        for (int nt = 0; nt < NNT; ++nt) {
            acc[0][nt] = __builtin_amdgcn_wmma_f32_16x16x32_bf16(
                false, a0, false, b[nt], (short)0, acc[0][nt], false, false);
            acc[1][nt] = __builtin_amdgcn_wmma_f32_16x16x32_bf16(
                false, a1, false, b[nt], (short)0, acc[1][nt], false, false);
        }
    }

    // ---- epilogue: bias + eval-BN + ReLU + residual, in-place h update ----
#pragma unroll
    for (int nt = 0; nt < NNT; ++nt) {
        int n = nt * 16 + mln;
        if (n < DD) {
            float gm = gamma[layer * DD + n] * INV_BN;
            float bt = beta[layer * DD + n];
            float bb = post_b[layer * DD + n];
#pragma unroll
            for (int t = 0; t < 2; ++t) {
#pragma unroll
                for (int i = 0; i < 8; ++i) {
                    int node = node0 + t * 16 + 8 * half + i;
                    if (node < NN) {
                        float cv = acc[t][nt][i] + bb;
                        float o  = fmaxf(cv * gm + bt, 0.0f) + h[node * DD + n];
                        h[node * DD + n] = o;
                    }
                }
            }
        }
    }
}

// ---------------- global mean pool ----------------
__global__ void k_pool(const int* __restrict__ batch, const float* __restrict__ h,
                       float* __restrict__ gsum) {
    for (int id = blockIdx.x * blockDim.x + threadIdx.x; id < NN * DD;
         id += gridDim.x * blockDim.x) {
        int n = id / DD, d = id - n * DD;
        atomicAdd(&gsum[batch[n] * DD + d], h[id]);
    }
}

__global__ void k_gcnt(const int* __restrict__ batch, float* __restrict__ gcnt) {
    for (int n = blockIdx.x * blockDim.x + threadIdx.x; n < NN; n += gridDim.x * blockDim.x)
        atomicAdd(&gcnt[batch[n]], 1.0f);
}

// ---------------- tiny MLP head: 70 -> 35 -> 17 -> 1 ----------------
__global__ void k_mlp(const float* __restrict__ gsum, const float* __restrict__ gcnt,
                      const float* __restrict__ w1, const float* __restrict__ b1,
                      const float* __restrict__ w2, const float* __restrict__ b2,
                      const float* __restrict__ w3, const float* __restrict__ b3,
                      float* __restrict__ out) {
    int g = blockIdx.x * blockDim.x + threadIdx.x;
    if (g >= NG) return;
    float inv = 1.0f / fmaxf(gcnt[g], 1.0f);
    float gv[DD];
#pragma unroll 1
    for (int d = 0; d < DD; ++d) gv[d] = gsum[g * DD + d] * inv;
    float t1[35];
#pragma unroll 1
    for (int j = 0; j < 35; ++j) {
        float a = b1[j];
        for (int d = 0; d < DD; ++d) a += gv[d] * w1[d * 35 + j];
        t1[j] = fmaxf(a, 0.0f);
    }
    float t2[17];
#pragma unroll 1
    for (int k = 0; k < 17; ++k) {
        float a = b2[k];
        for (int j = 0; j < 35; ++j) a += t1[j] * w2[k + j * 17];
        t2[k] = fmaxf(a, 0.0f);
    }
    float a = b3[0];
    for (int k = 0; k < 17; ++k) a += t2[k] * w3[k];
    out[g] = a;
}

// ---------------- launcher ----------------
extern "C" void kernel_launch(void* const* d_in, const int* in_sizes, int n_in,
                              void* d_out, int out_size, void* d_ws, size_t ws_size,
                              hipStream_t stream) {
    const int*   x       = (const int*)d_in[0];
    const int*   ei      = (const int*)d_in[1];
    const float* ea      = (const float*)d_in[2];
    const int*   batch   = (const int*)d_in[3];
    const float* emb     = (const float*)d_in[4];
    const float* post_w  = (const float*)d_in[5];
    const float* post_b  = (const float*)d_in[6];
    const float* gamma   = (const float*)d_in[7];
    const float* beta    = (const float*)d_in[8];
    const float* w1      = (const float*)d_in[9];
    const float* b1      = (const float*)d_in[10];
    const float* w2      = (const float*)d_in[11];
    const float* b2      = (const float*)d_in[12];
    const float* w3      = (const float*)d_in[13];
    const float* b3      = (const float*)d_in[14];
    float* outp = (float*)d_out;

    // workspace carve-out
    char* base = (char*)d_ws;
    size_t off = 0;
    auto take = [&](size_t bytes) -> char* {
        char* p = base + off;
        off = (off + bytes + 255) & ~(size_t)255;
        return p;
    };
    float*          h     = (float*)take((size_t)NN * DD * 4);
    float*          s     = (float*)take((size_t)NN * DD * 4);
    float*          s2    = (float*)take((size_t)NN * DD * 4);
    int*            mn    = (int*)take((size_t)NN * DD * 4);
    int*            mx    = (int*)take((size_t)NN * DD * 4);
    float*          cnt   = (float*)take((size_t)NN * 4);
    float*          amp   = (float*)take((size_t)NN * 4);
    unsigned short* wpack = (unsigned short*)take((size_t)NLAYER * WL_STRIDE * 2);
    float*          gsum  = (float*)take((size_t)NG * DD * 4);
    float*          gcnt  = (float*)take((size_t)NG * 4);
    (void)ws_size; (void)in_sizes; (void)n_in; (void)out_size;

    const int TB = 256;
    auto blocks = [&](long n) { return (int)((n + TB - 1) / TB); };

    // once-per-call prep (graph-replay safe: everything rebuilt each call)
    k_zero_f<<<blocks(NN), TB, 0, stream>>>(cnt, NN);
    k_zero_f<<<blocks((long)NG * DD), TB, 0, stream>>>(gsum, NG * DD);
    k_zero_f<<<blocks(NG), TB, 0, stream>>>(gcnt, NG);
    k_count<<<blocks(NE), TB, 0, stream>>>(ei, cnt);
    k_amp<<<blocks(NN), TB, 0, stream>>>(cnt, amp);
    k_gcnt<<<blocks(NN), TB, 0, stream>>>(batch, gcnt);
    k_wpack<<<blocks((long)NLAYER * WL_STRIDE), TB, 0, stream>>>(post_w, wpack);
    k_atom<<<blocks((long)NN * DD), TB, 0, stream>>>(x, emb, h);

    const int nodeBlocks = (NN + 31) / 32;   // one 32-node wave-tile per block
    for (int l = 0; l < NLAYER; ++l) {
        k_init_layer<<<blocks((long)NN * DD), TB, 0, stream>>>(s, s2, mn, mx, NN * DD);
        k_edge<<<blocks((long)NE * DD), TB, 0, stream>>>(ei, ea, h, s, s2, mn, mx);
        k_node<<<nodeBlocks, 32, 0, stream>>>(s, s2, mn, mx, cnt, amp, wpack,
                                              post_b, gamma, beta, h, l);
    }

    k_pool<<<blocks((long)NN * DD), TB, 0, stream>>>(batch, h, gsum);
    k_mlp<<<(NG + 63) / 64, 64, 0, stream>>>(gsum, gcnt, w1, b1, w2, b2, w3, b3, outp);
}
